// CNN_LSTM_ActorCritic_56049323213226
// MI455X (gfx1250) — matrix operand order repaired and checked by reading, not compiled, not run on hardware
//
#include <hip/hip_runtime.h>

typedef _Float16 v16h __attribute__((ext_vector_type(16)));
typedef _Float16 v8h  __attribute__((ext_vector_type(8)));
typedef float    v8f  __attribute__((ext_vector_type(8)));
typedef unsigned int ui32x4 __attribute__((ext_vector_type(4)));
typedef int          i32x8  __attribute__((ext_vector_type(8)));
typedef int          i32x4  __attribute__((ext_vector_type(4)));

#define DEV static __device__ __forceinline__

DEV v16h cat8(v8h lo, v8h hi){
  return __builtin_shufflevector(lo, hi, 0,1,2,3,4,5,6,7,8,9,10,11,12,13,14,15);
}
DEV v8f wmma_f16(v16h a, v16h b, v8f c){
  // D = A(16x32 f16) * B(32x16 f16) + C(16x16 f32)
  return __builtin_amdgcn_wmma_f32_16x16x32_f16(false, a, false, b, (short)0, c, false, false);
}
// A fragment from a K-contiguous f16 row (lane layout: e<8 -> k=half*8+e ; e>=8 -> k=16+half*8+(e-8))
DEV v16h load_afrag_row(const _Float16* __restrict__ arow, int kt, int half){
  const _Float16* q = arow + kt*32 + half*8;
  return cat8(*(const v8h*)q, *(const v8h*)(q + 16));
}
// B fragment from [N][K] row-major f16 (lane layout: e<8 -> k=half*8+e ; e>=8 -> k=16+half*8+(e-8))
DEV v16h load_bfrag(const _Float16* __restrict__ Bf, int K, int col, int kt, int half){
  const _Float16* p = Bf + (size_t)col*K + kt*32 + half*8;
  return cat8(*(const v8h*)p, *(const v8h*)(p + 16));
}
DEV float sigmoidf_(float x){ return 1.0f/(1.0f + __expf(-x)); }

// ---- Tensor Data Mover: DMA L contiguous f16 elements from global into LDS ----
// D# per CDNA5 ISA: group0 = {count=1 | lds_addr | global_addr | type=2},
// group1 = {data_size=2B, tensor_dim0=tile_dim0=L (one row), tensor_dim1=1, stride0=L}.
// This toolchain exposes the 6-arg builtin: (ui32x4, i32x8, i32x4, i32x4, i32x8, i32 cpol).
DEV void tdm_load_1d_f16(unsigned lds_byte, const void* gptr, unsigned L){
  unsigned long long ga = (unsigned long long)(size_t)gptr;
  ui32x4 g0 = { 1u,                                    // count=1, user mode
                lds_byte,                              // lds_addr (bytes)
                (unsigned)ga,                          // global_addr[31:0]
                (unsigned)((ga >> 32) & 0x1FFFFFFull)  // global_addr[56:32]
                  | (2u << 30) };                      // type=2 ("image")
  i32x8 g1 = { (int)(1u << 16),                        // data_size=1 (2 bytes), mask=0
               (int)((L & 0xFFFFu) << 16),             // tensor_dim0[15:0]
               (int)((L >> 16) | (1u << 16)),          // tensor_dim0[31:16] | tensor_dim1=1
               (int)((L & 0xFFFFu) << 16),             // tile_dim0 = L
               0,                                      // tile_dim1=0, tile_dim2=0
               (int)L,                                 // tensor_dim0_stride[31:0]
               0, 0 };
  i32x4 gz4 = {0,0,0,0};
  i32x8 gz8 = {0,0,0,0,0,0,0,0};
  __builtin_amdgcn_tensor_load_to_lds(g0, g1, gz4, gz4, gz8, 0);
}

// ---------------------------------------------------------------- prep kernels
__global__ void cvt_f16_kernel(const float* __restrict__ s, _Float16* __restrict__ d, int n){
  int i = blockIdx.x*256 + threadIdx.x;
  if (i < n) d[i] = (_Float16)s[i];
}
__global__ void copy_f32_kernel(const float* __restrict__ s, float* __restrict__ d, int n){
  int i = blockIdx.x*256 + threadIdx.x;
  if (i < n) d[i] = s[i];
}
// conv weights OIHW -> f16 [OC][K], k ordered (kh,kw,ic) to match NHWC activations
__global__ void pack_conv_w_nhwc(const float* __restrict__ w, _Float16* __restrict__ d,
                                 int OC, int IC, int KH, int KW){
  int K = IC*KH*KW;
  int i = blockIdx.x*256 + threadIdx.x;
  if (i >= OC*K) return;
  int oc = i / K, k = i % K;
  int khkw = k / IC, ic = k % IC;
  d[i] = (_Float16)w[((size_t)oc*IC + ic)*(KH*KW) + khkw];
}
// fc_w [512][4608] NCHW-flatten cols -> f16 [512][4608] with k in NHWC order
__global__ void pack_fc_w_kernel(const float* __restrict__ w, _Float16* __restrict__ d){
  int i = blockIdx.x*256 + threadIdx.x;
  if (i >= 512*4608) return;
  int n = i / 4608, k = i % 4608;
  int c = k & 31, hw = k >> 5;                 // k = (h*12+w)*32 + c
  d[i] = (_Float16)w[(size_t)n*4608 + c*144 + hw];
}

// ---------------------------------------------------------------- implicit-GEMM conv (weights TDM-staged in LDS)
template<int IC,int IH,int IW,int OC,int OH,int OW,int KH,int KW,int S,bool IN_NCHW_F32>
__global__ __launch_bounds__(256)
void conv_wmma_kernel(const void* __restrict__ in_, const _Float16* __restrict__ wf,
                      const float* __restrict__ bias, _Float16* __restrict__ out){
  constexpr int K  = IC*KH*KW;
  constexpr int KT = K/32;
  constexpr int NT = OC/16;
  extern __shared__ _Float16 smem[];            // [OC][K] f16 weight panel

  if ((threadIdx.x >> 5) == 0){                 // wave 0 issues the DMA
    tdm_load_1d_f16((unsigned)(size_t)&smem[0], wf, (unsigned)(OC*K));
    __builtin_amdgcn_s_wait_tensorcnt(0);
  }
  __syncthreads();

  const int lane  = threadIdx.x & 31;
  const int wave  = threadIdx.x >> 5;
  const int half  = lane >> 4;
  const int mr    = lane & 15;
  const int mtile = blockIdx.x*8 + wave;
  const int mA  = mtile*16 + mr;                // output pixel this lane gathers for
  const int img = mA/(OH*OW);
  const int pix = mA%(OH*OW);
  const int oh  = pix/OW, ow = pix%OW;
  const int ih0 = oh*S,   iw0 = ow*S;

  v8f zero = {};
  v8f acc[NT];
  #pragma unroll
  for (int n=0;n<NT;n++) acc[n] = zero;

  for (int kt=0; kt<KT; ++kt){
    v16h a;
    if constexpr (IN_NCHW_F32){
      const float* in = (const float*)in_;
      #pragma unroll
      for (int e=0;e<16;e++){
        int k  = kt*32 + ((e>>3)<<4) + half*8 + (e&7);   // k ordered (ic,kh,kw)
        int ic = k/(KH*KW); int r2 = k%(KH*KW);
        int kh = r2/KW;     int kw = r2 - kh*KW;
        float v = in[(((size_t)img*IC + ic)*IH + (ih0+kh))*IW + (iw0+kw)];
        a[e] = (_Float16)(v * (1.0f/255.0f));
      }
    } else {
      const _Float16* in = (const _Float16*)in_;         // NHWC, k ordered (kh,kw,ic)
      v8h runs[2];
      #pragma unroll
      for (int r=0;r<2;r++){
        int k8   = kt*32 + r*16 + half*8;                // IC % 8 == 0 -> contiguous run
        int khkw = k8/IC; int ic0 = k8 - khkw*IC;
        int kh   = khkw/KW; int kw = khkw - kh*KW;
        const _Float16* p = in + (((size_t)img*IH + (ih0+kh))*IW + (iw0+kw))*IC + ic0;
        runs[r] = *(const v8h*)p;
      }
      a = cat8(runs[0], runs[1]);
    }
    #pragma unroll
    for (int n=0;n<NT;n++){
      const _Float16* bp = &smem[(size_t)(n*16 + mr)*K + kt*32 + half*8];
      v16h b = cat8(*(const v8h*)bp, *(const v8h*)(bp + 16));   // ds_load_b128 x2
      acc[n] = wmma_f16(a, b, acc[n]);
    }
  }
  const int mbase = mtile*16 + half*8;          // C row = output pixel, C col = mr
  #pragma unroll
  for (int n=0;n<NT;n++){
    const int col = n*16 + mr;
    const float bc = bias[col];
    #pragma unroll
    for (int v=0;v<8;v++){
      float val = acc[n][v] + bc;
      val = val > 0.0f ? val : 0.0f;                           // ReLU
      out[(size_t)(mbase+v)*OC + col] = (_Float16)val;         // NHWC
    }
  }
}

// ---------------------------------------------------------------- WMMA GEMM, 32x64 per wave (2 M-tiles x 4 N-tiles)
__global__ __launch_bounds__(256)
void gemm_wmma_kernel(const _Float16* __restrict__ A, const _Float16* __restrict__ Bf,
                      const float* __restrict__ bias, int N, int K, int ngroups,
                      float* __restrict__ outF32, _Float16* __restrict__ outF16, int relu){
  const int lane = threadIdx.x & 31, wave = threadIdx.x >> 5;
  const int wid  = blockIdx.x*8 + wave;
  const int mpair = wid / ngroups, ng = wid % ngroups;
  const int half = lane >> 4, mr = lane & 15;
  const _Float16* arow0 = A + (size_t)(mpair*32 + mr)*K;
  const _Float16* arow1 = arow0 + (size_t)16*K;
  v8f acc[2][4] = {};
  const int KT = K/32;
  for (int kt=0; kt<KT; ++kt){
    if (kt + 2 < KT){
      __builtin_prefetch(arow0 + (kt+2)*32, 0, 3);   // global_prefetch_b8 (near)
      __builtin_prefetch(arow1 + (kt+2)*32, 0, 3);
    }
    v16h a0 = load_afrag_row(arow0, kt, half);
    v16h a1 = load_afrag_row(arow1, kt, half);
    #pragma unroll
    for (int j=0;j<4;j++){
      v16h b = load_bfrag(Bf, K, (ng*4+j)*16 + mr, kt, half);  // B reused by both M-tiles
      acc[0][j] = wmma_f16(a0, b, acc[0][j]);
      acc[1][j] = wmma_f16(a1, b, acc[1][j]);
    }
  }
  #pragma unroll
  for (int mi=0; mi<2; ++mi){
    const int rbase = mpair*32 + mi*16 + half*8;
    #pragma unroll
    for (int j=0;j<4;j++){
      int col = (ng*4+j)*16 + mr;
      float bc = bias ? bias[col] : 0.0f;
      #pragma unroll
      for (int v=0;v<8;v++){
        float val = acc[mi][j][v] + bc;
        if (relu) val = fmaxf(val, 0.0f);
        if (outF32) outF32[(size_t)(rbase+v)*N + col] = val;
        if (outF16) outF16[(size_t)(rbase+v)*N + col] = (_Float16)val;
      }
    }
  }
}

// ---------------------------------------------------------------- one GRU step (M=16,K=512; 16 hidden cols/block, 3 gates)
// Single-wave block; the 3 gate weight strips (48KB) are TDM-staged into LDS.
__global__ __launch_bounds__(32)
void gru_step_kernel(const float* __restrict__ h_in, const float* __restrict__ gi,
                     const float* __restrict__ masks, const _Float16* __restrict__ whh,
                     const float* __restrict__ b_hh, int t,
                     float* __restrict__ h_out, float* __restrict__ xout){
  extern __shared__ _Float16 wsm[];              // [3][16][512] f16
  const int lane = threadIdx.x;
  const int jt   = blockIdx.x;                   // hidden cols jt*16..+15
  const int half = lane >> 4, mr = lane & 15;

  unsigned lds0 = (unsigned)(size_t)&wsm[0];
  #pragma unroll
  for (int g=0; g<3; ++g)                        // 16 rows x 512 K, contiguous in [N][K]
    tdm_load_1d_f16(lds0 + (unsigned)g*16384u,
                    whh + ((size_t)g*512 + jt*16)*512, 16*512);
  __builtin_amdgcn_s_wait_tensorcnt(0);          // single wave: no barrier needed

  const float mk = masks[t*16 + mr];             // A-side batch mask
  const float* hrow = h_in + mr*512;
  v8f aR = {}, aZ = {}, aN = {};
  for (int kt=0; kt<16; ++kt){
    v16h a;
    #pragma unroll
    for (int e=0;e<16;e++){
      int k = kt*32 + ((e>>3)<<4) + half*8 + (e&7);
      a[e] = (_Float16)(hrow[k] * mk);           // mask-reset folded into A
    }
    #pragma unroll
    for (int g=0; g<3; ++g){
      const _Float16* bp = &wsm[(size_t)g*8192 + mr*512 + kt*32 + half*8];
      v16h b = cat8(*(const v8h*)bp, *(const v8h*)(bp + 16));
      if (g==0) aR = wmma_f16(a, b, aR);
      else if (g==1) aZ = wmma_f16(a, b, aZ);
      else aN = wmma_f16(a, b, aN);
    }
  }
  const int j = jt*16 + mr;                      // C-side hidden column
  const float bhr = b_hh[j], bhz = b_hh[512+j], bhn = b_hh[1024+j];
  #pragma unroll
  for (int v=0;v<8;v++){
    int m   = half*8 + v;                        // C-side batch row
    int row = t*16 + m;
    float hp = h_in[m*512 + j] * masks[row];
    float r  = sigmoidf_(gi[(size_t)row*1536 +        j] + aR[v] + bhr);
    float z  = sigmoidf_(gi[(size_t)row*1536 +  512 + j] + aZ[v] + bhz);
    float nn = tanhf(    gi[(size_t)row*1536 + 1024 + j] + r*(aN[v] + bhn));
    float hn = (1.0f - z)*nn + z*hp;
    h_out[m*512 + j] = hn;
    xout[(size_t)row*512 + j] = hn;
  }
}

// ---------------------------------------------------------------- actor/critic heads
__global__ __launch_bounds__(32)
void heads_kernel(const float* __restrict__ xout, const float* __restrict__ aw,
                  const float* __restrict__ ab, const float* __restrict__ cw,
                  const float* __restrict__ cb, const int* __restrict__ action,
                  float* __restrict__ out){
  const int row = blockIdx.x, lane = threadIdx.x;
  const float* xr = xout + (size_t)row*512;
  float dots[7];
  #pragma unroll
  for (int a=0;a<7;a++){
    const float* w = (a<6) ? (aw + a*512) : cw;
    float s = 0.0f;
    for (int k=lane; k<512; k+=32) s += xr[k]*w[k];
    #pragma unroll
    for (int off=16; off>0; off>>=1) s += __shfl_xor(s, off, 32);
    dots[a] = s;
  }
  if (lane == 0){
    float logits[6];
    float mx = -1e30f;
    #pragma unroll
    for (int a=0;a<6;a++){ logits[a] = dots[a] + ab[a]; mx = fmaxf(mx, logits[a]); }
    float se = 0.0f;
    #pragma unroll
    for (int a=0;a<6;a++) se += __expf(logits[a] - mx);
    float lse = mx + __logf(se);
    float ent = 0.0f;
    #pragma unroll
    for (int a=0;a<6;a++){ float lp = logits[a] - lse; ent -= __expf(lp)*lp; }
    int act = action[row];
    out[row]        = dots[6] + cb[0];           // value
    out[512 + row]  = logits[act] - lse;         // action_log_probs
    out[1024 + row] = ent;                       // entropy
  }
}

// ---------------------------------------------------------------- host
extern "C" void kernel_launch(void* const* d_in, const int* in_sizes, int n_in,
                              void* d_out, int out_size, void* d_ws, size_t ws_size,
                              hipStream_t stream){
  (void)in_sizes; (void)n_in; (void)out_size; (void)ws_size;
  const float* inputs = (const float*)d_in[0];
  const float* states = (const float*)d_in[1];
  const float* masks  = (const float*)d_in[2];
  const int*   action = (const int*)  d_in[3];
  const float* c1w = (const float*)d_in[4];  const float* c1b = (const float*)d_in[5];
  const float* c2w = (const float*)d_in[6];  const float* c2b = (const float*)d_in[7];
  const float* c3w = (const float*)d_in[8];  const float* c3b = (const float*)d_in[9];
  const float* fcw = (const float*)d_in[10]; const float* fcb = (const float*)d_in[11];
  const float* wih = (const float*)d_in[12]; const float* whh = (const float*)d_in[13];
  const float* bih = (const float*)d_in[14]; const float* bhh = (const float*)d_in[15];
  const float* aw  = (const float*)d_in[16]; const float* ab  = (const float*)d_in[17];
  const float* cw  = (const float*)d_in[18]; const float* cb  = (const float*)d_in[19];
  float* out = (float*)d_out;

  char* ws = (char*)d_ws;
  size_t off = 0;
  auto take = [&](size_t bytes)->char*{
    char* p = ws + off; off += (bytes + 255) & ~(size_t)255; return p;
  };
  _Float16* w1f  = (_Float16*)take((size_t)32*256*2);
  _Float16* w2f  = (_Float16*)take((size_t)64*512*2);
  _Float16* w3f  = (_Float16*)take((size_t)32*576*2);
  _Float16* fcf  = (_Float16*)take((size_t)512*4608*2);
  _Float16* wihf = (_Float16*)take((size_t)1536*512*2);
  _Float16* whhf = (_Float16*)take((size_t)1536*512*2);
  _Float16* a1   = (_Float16*)take((size_t)512*30*30*32*2);
  _Float16* a2   = (_Float16*)take((size_t)512*14*14*64*2);
  _Float16* a3   = (_Float16*)take((size_t)512*12*12*32*2);
  _Float16* x16  = (_Float16*)take((size_t)512*512*2);
  float*    gi   = (float*)   take((size_t)512*1536*4);
  float*    hA   = (float*)   take((size_t)16*512*4);
  float*    hB   = (float*)   take((size_t)16*512*4);
  float*    xo   = (float*)   take((size_t)512*512*4);

  // weight prep (f16 convert + layout permutes)
  cvt_f16_kernel   <<<(32*256   +255)/256,256,0,stream>>>(c1w, w1f, 32*256);
  pack_conv_w_nhwc <<<(64*512   +255)/256,256,0,stream>>>(c2w, w2f, 64,32,4,4);
  pack_conv_w_nhwc <<<(32*576   +255)/256,256,0,stream>>>(c3w, w3f, 32,64,3,3);
  pack_fc_w_kernel <<<(512*4608 +255)/256,256,0,stream>>>(fcw, fcf);
  cvt_f16_kernel   <<<(1536*512 +255)/256,256,0,stream>>>(wih, wihf, 1536*512);
  cvt_f16_kernel   <<<(1536*512 +255)/256,256,0,stream>>>(whh, whhf, 1536*512);
  copy_f32_kernel  <<<(16*512   +255)/256,256,0,stream>>>(states, hA, 16*512);

  // CNN trunk (implicit-GEMM WMMA, NHWC f16 activations, TDM-staged weights in LDS)
  conv_wmma_kernel< 4,124,124,32,30,30,8,8,4,true ><<<3600,256,16384,stream>>>(inputs, w1f, c1b, a1);
  conv_wmma_kernel<32, 30, 30,64,14,14,4,4,2,false><<< 784,256,65536,stream>>>(a1,     w2f, c2b, a2);
  conv_wmma_kernel<64, 14, 14,32,12,12,3,3,1,false><<< 576,256,36864,stream>>>(a2,     w3f, c3b, a3);

  // FC: [512,4608]x[4608,512] -> f16 (ReLU);  gi: [512,512]x[512,1536] -> f32 (+b_ih)
  gemm_wmma_kernel<<<16,256,0,stream>>>(a3,  fcf,  fcb,  512, 4608,  8, nullptr, x16, 1);
  gemm_wmma_kernel<<<48,256,0,stream>>>(x16, wihf, bih, 1536,  512, 24, gi,  nullptr, 0);

  // GRU scan: 32 sequential steps, double-buffered h; final h lands in d_out states slot
  float* hin = hA; float* hout = hB;
  for (int t=0; t<32; ++t){
    float* ho = (t == 31) ? (out + 1536) : hout;
    gru_step_kernel<<<32,32,49152,stream>>>(hin, gi, masks, whhf, bhh, t, ho, xo);
    float* tmp = hin; hin = hout; hout = tmp;
  }

  // heads: value / action_log_probs / entropy
  heads_kernel<<<512,32,0,stream>>>(xo, aw, ab, cw, cb, action, out);
}
